// SwinTransformer3D_34205119545831
// MI455X (gfx1250) — matrix operand order repaired and checked by reading, not compiled, not run on hardware
//
#include <hip/hip_runtime.h>
#include <hip/hip_fp16.h>
#include <hip/hip_bf16.h>

// ---------------------------------------------------------------------------
// Swin-3D block for MI455X (gfx1250): wave32 + v_wmma_f32_16x16x32_f16.
// All GEMMs run in f16 with f32 accumulation; epilogues fuse bias/scale/GELU/
// residual and the window-partition / cyclic-roll index remaps so every
// intermediate touches HBM exactly once.  The streaming A operand of each
// GEMM is double-buffered into LDS with async global->LDS copies (ASYNCcnt)
// when the toolchain exposes the gfx1250 builtins.
// ---------------------------------------------------------------------------

typedef __attribute__((ext_vector_type(16))) _Float16 v16h;
typedef __attribute__((ext_vector_type(8)))  _Float16 v8h;
typedef __attribute__((ext_vector_type(8)))  float    v8f;
typedef int i32x4 __attribute__((vector_size(16)));   // async-copy operand type

#define NUM_HEADS 12
#define C_DIM     384

#if defined(__gfx1250__) &&                                                  \
    __has_builtin(__builtin_amdgcn_global_load_async_to_lds_b128) &&         \
    __has_builtin(__builtin_amdgcn_s_wait_asynccnt)
#define HAVE_ASYNC 1
#else
#define HAVE_ASYNC 0
#endif

__device__ __forceinline__ v16h concat8(v8h a, v8h b) {
  return __builtin_shufflevector(a, b, 0,1,2,3,4,5,6,7,8,9,10,11,12,13,14,15);
}

// A-matrix fragment, 16x32 f16 (MxK), source row-major with leading dim ld.
// ISA layout: lanes 0-15 hold row M=lane, K = 0..7 (v0..v3) and 16..23 (v4..v7);
// lanes 16-31 hold K = 8..15 and 24..31.  => two contiguous 16B loads per lane.
__device__ __forceinline__ v16h load_a_frag(const _Float16* base, int ld, int lane) {
  int row  = lane & 15;
  int koff = (lane >> 4) << 3;          // 0 or 8
  const _Float16* p = base + (size_t)row * ld + koff;
  v8h lo = *(const v8h*)(p);            // K = koff .. koff+7
  v8h hi = *(const v8h*)(p + 16);       // K = koff+16 .. koff+23
  return concat8(lo, hi);
}

// B-matrix fragment, 32x16 f16 (KxN), given Bt = B^T (N rows, K cols) row-major.
// Layout matching A: lanes 0-15 hold K=0..7 and 16..23 of column N=lane,
// lanes 16-31 hold K=8..15 and 24..31.  => two contiguous 16B loads per lane.
__device__ __forceinline__ v16h load_b_frag(const _Float16* base, int ld, int lane) {
  int col  = lane & 15;
  int koff = (lane >> 4) << 3;          // 0 or 8
  const _Float16* p = base + (size_t)col * ld + koff;
  v8h lo = *(const v8h*)(p);
  v8h hi = *(const v8h*)(p + 16);
  return concat8(lo, hi);
}

__device__ __forceinline__ v8f wmma_f32(v16h a, v16h b, v8f c) {
  return __builtin_amdgcn_wmma_f32_16x16x32_f16(false, a, false, b, (short)0, c,
                                                false, false);
}

__device__ __forceinline__ int reg3(int g) {  // shifted-coord region: [0,28) [28,30) [30,32)
  return (g < 28) ? 0 : ((g < 30) ? 1 : 2);
}

// ---------------------------------------------------------------------------
__global__ void f32_to_f16_kernel(const float* __restrict__ s,
                                  _Float16* __restrict__ d, int n) {
  int i = blockIdx.x * 256 + threadIdx.x;
  if (i < n) d[i] = (_Float16)s[i];
}

// LN1 fused with cyclic shift (-2) and 4x4x4 window partition. One wave per row.
__global__ __launch_bounds__(256)
void ln1_shift_partition(const float* __restrict__ x, const float* __restrict__ g,
                         const float* __restrict__ b, _Float16* __restrict__ xw) {
  int row  = blockIdx.x * 8 + (threadIdx.x >> 5);   // natural token index
  int lane = threadIdx.x & 31;
  const float* xr = x + (size_t)row * C_DIM;
  float v[12]; float s = 0.f;
#pragma unroll
  for (int i = 0; i < 12; ++i) { v[i] = xr[lane + 32 * i]; s += v[i]; }
#pragma unroll
  for (int o = 16; o > 0; o >>= 1) s += __shfl_xor(s, o, 32);
  float mu = s * (1.f / 384.f);
  float var = 0.f;
#pragma unroll
  for (int i = 0; i < 12; ++i) { float d = v[i] - mu; var += d * d; }
#pragma unroll
  for (int o = 16; o > 0; o >>= 1) var += __shfl_xor(var, o, 32);
  float rstd = rsqrtf(var * (1.f / 384.f) + 1e-5f);

  int bb = row >> 15; int sp = row & 32767;
  int gd = ((sp >> 10) + 30) & 31;        // shifted coord = orig - 2 (mod 32)
  int gh = (((sp >> 5) & 31) + 30) & 31;
  int gw = ((sp & 31) + 30) & 31;
  int b_  = ((bb * 8 + (gd >> 2)) * 8 + (gh >> 2)) * 8 + (gw >> 2);
  int tok = (gd & 3) * 16 + (gh & 3) * 4 + (gw & 3);
  _Float16* orow = xw + ((size_t)b_ * 64 + tok) * C_DIM;
#pragma unroll
  for (int i = 0; i < 12; ++i) {
    int c = lane + 32 * i;
    orow[c] = (_Float16)((v[i] - mu) * rstd * g[c] + b[c]);
  }
}

// Plain LN (for the MLP branch), natural layout, f16 output.
__global__ __launch_bounds__(256)
void ln2_kernel(const float* __restrict__ x, const float* __restrict__ g,
                const float* __restrict__ b, _Float16* __restrict__ h2) {
  int row  = blockIdx.x * 8 + (threadIdx.x >> 5);
  int lane = threadIdx.x & 31;
  const float* xr = x + (size_t)row * C_DIM;
  float v[12]; float s = 0.f;
#pragma unroll
  for (int i = 0; i < 12; ++i) { v[i] = xr[lane + 32 * i]; s += v[i]; }
#pragma unroll
  for (int o = 16; o > 0; o >>= 1) s += __shfl_xor(s, o, 32);
  float mu = s * (1.f / 384.f);
  float var = 0.f;
#pragma unroll
  for (int i = 0; i < 12; ++i) { float d = v[i] - mu; var += d * d; }
#pragma unroll
  for (int o = 16; o > 0; o >>= 1) var += __shfl_xor(var, o, 32);
  float rstd = rsqrtf(var * (1.f / 384.f) + 1e-5f);
  _Float16* orow = h2 + (size_t)row * C_DIM;
#pragma unroll
  for (int i = 0; i < 12; ++i) {
    int c = lane + 32 * i;
    orow[c] = (_Float16)((v[i] - mu) * rstd * g[c] + b[c]);
  }
}

// ---------------------------------------------------------------------------
// Generic WMMA GEMM: C(MxN) = A(MxK,f16 row-major) * Bt(NxK,f16 row-major)^T.
// Block: 256 threads = 8 waves; block tile 128(M) x 64(N); wave tile 32x32.
// A is staged through a double-buffered LDS tile (padded stride to kill bank
// conflicts) with async global->LDS copies; B (L2-resident weights) is read
// directly as fragments.
// MODE 0: QKV scatter (q scaled, k token-major, v transposed to hd x N)
// MODE 1: proj: + bias, reverse-partition + roll(+2) remap, + shortcut -> f32
// MODE 2: MLP1: + bias, exact GELU -> f16
// MODE 3: MLP2: + bias, + residual -> f32
// ---------------------------------------------------------------------------
#define A_TS 40   // LDS tile stride in halves (32 data + 8 pad = 80B rows)

template <int MODE>
__global__ __launch_bounds__(256)
void gemm_wmma_kernel(const _Float16* __restrict__ A, const _Float16* __restrict__ Bt,
                      const float* __restrict__ bias, int M, int N, int K,
                      _Float16* __restrict__ o16q, _Float16* __restrict__ o16k,
                      _Float16* __restrict__ o16v, const float* __restrict__ res,
                      float* __restrict__ o32) {
  __shared__ alignas(16) _Float16 Atile[2][128 * A_TS];
  int tid  = threadIdx.x;
  int lane = tid & 31;
  int wave = tid >> 5;
  int m0 = blockIdx.y * 128 + (wave >> 1) * 32;
  int n0 = blockIdx.x * 64  + (wave & 1) * 32;

  const _Float16* Ablk = A  + (size_t)blockIdx.y * 128 * K;
  const _Float16* Bb   = Bt + (size_t)n0 * K;

  // Cooperative A staging: 128x32 f16 tile = 512 chunks of 16B; 2 per thread.
  int cr = tid >> 2;        // chunk row 0..63 (second chunk: +64)
  int cs = tid & 3;         // 16B segment within the 64B row slice

  auto stage = [&](int kb, int buf) {
    const _Float16* gp0 = Ablk + (size_t)cr * K + kb + cs * 8;
    const _Float16* gp1 = Ablk + (size_t)(cr + 64) * K + kb + cs * 8;
    _Float16* lp0 = &Atile[buf][cr * A_TS + cs * 8];
    _Float16* lp1 = &Atile[buf][(cr + 64) * A_TS + cs * 8];
#if HAVE_ASYNC
    __builtin_amdgcn_global_load_async_to_lds_b128(
        (__attribute__((address_space(1))) i32x4*)(_Float16*)gp0,
        (__attribute__((address_space(3))) i32x4*)lp0, 0, 0);
    __builtin_amdgcn_global_load_async_to_lds_b128(
        (__attribute__((address_space(1))) i32x4*)(_Float16*)gp1,
        (__attribute__((address_space(3))) i32x4*)lp1, 0, 0);
#else
    *(v8h*)lp0 = *(const v8h*)gp0;
    *(v8h*)lp1 = *(const v8h*)gp1;
#endif
  };

  v8f acc[2][2];
#pragma unroll
  for (int i = 0; i < 2; ++i)
#pragma unroll
    for (int j = 0; j < 2; ++j) acc[i][j] = (v8f){0.f,0.f,0.f,0.f,0.f,0.f,0.f,0.f};

  int nk = K / 32;
  stage(0, 0);
  for (int s = 0; s < nk; ++s) {
    if (s + 1 < nk) {
      stage((s + 1) * 32, (s + 1) & 1);   // prefetch next k-step
#if HAVE_ASYNC
      __builtin_amdgcn_s_wait_asynccnt(2);  // retire previous stage's 2 copies
#endif
    } else {
#if HAVE_ASYNC
      __builtin_amdgcn_s_wait_asynccnt(0);
#endif
    }
    __syncthreads();                       // staged tile visible to all waves

    const _Float16* At = &Atile[s & 1][((wave >> 1) * 32) * A_TS];
    v16h a0 = load_a_frag(At, A_TS, lane);
    v16h a1 = load_a_frag(At + 16 * A_TS, A_TS, lane);
    v16h b0 = load_b_frag(Bb + s * 32, K, lane);
    v16h b1 = load_b_frag(Bb + (size_t)16 * K + s * 32, K, lane);
    acc[0][0] = wmma_f32(a0, b0, acc[0][0]);
    acc[0][1] = wmma_f32(a0, b1, acc[0][1]);
    acc[1][0] = wmma_f32(a1, b0, acc[1][0]);
    acc[1][1] = wmma_f32(a1, b1, acc[1][1]);
    __syncthreads();                       // reads done before buffer reuse
  }

  int colL = lane & 15;
  int rsel = (lane >> 4) << 3;
#pragma unroll
  for (int mi = 0; mi < 2; ++mi) {
#pragma unroll
    for (int ni = 0; ni < 2; ++ni) {
#pragma unroll
      for (int e = 0; e < 8; ++e) {
        int row = m0 + mi * 16 + rsel + e;
        int col = n0 + ni * 16 + colL;
        float val = acc[mi][ni][e] + bias[col];
        if (MODE == 0) {
          int s  = col / C_DIM;                 // 0=q 1=k 2=v
          int r  = col - s * C_DIM;
          int hh = r >> 5;
          int dd = r & 31;
          int b_ = row >> 6;
          int tok = row & 63;
          if (s == 0)
            o16q[(((size_t)b_ * NUM_HEADS + hh) * 64 + tok) * 32 + dd] =
                (_Float16)(val * 0.17677669529663687f);   // hd^-0.5
          else if (s == 1)
            o16k[(((size_t)b_ * NUM_HEADS + hh) * 64 + tok) * 32 + dd] = (_Float16)val;
          else
            o16v[(((size_t)b_ * NUM_HEADS + hh) * 32 + dd) * 64 + tok] = (_Float16)val;
        } else if (MODE == 1) {
          int b_  = row >> 6; int tok = row & 63;
          int bb  = b_ >> 9;  int wf  = b_ & 511;
          int gd = ((wf >> 6) & 7) * 4 + (tok >> 4);
          int gh = ((wf >> 3) & 7) * 4 + ((tok >> 2) & 3);
          int gw = (wf & 7) * 4 + (tok & 3);
          int od = (gd + 2) & 31, oh = (gh + 2) & 31, ow = (gw + 2) & 31;
          size_t idx = ((((size_t)bb * 32 + od) * 32 + oh) * 32 + ow) * C_DIM + col;
          o32[idx] = res[idx] + val;           // shortcut + attn branch
        } else if (MODE == 2) {
          float gx = 0.5f * val * (1.0f + erff(val * 0.70710678118654752f));
          o16q[(size_t)row * N + col] = (_Float16)gx;
        } else {
          size_t idx = (size_t)row * N + col;
          o32[idx] = res[idx] + val;           // x + mlp branch
        }
      }
    }
  }
}

// ---------------------------------------------------------------------------
// Window attention: one wave per (window, head).  S = qk^T via K=32 WMMA,
// bias + shift-mask + softmax in LDS, then O = P·V via two K=32 WMMA steps.
// ---------------------------------------------------------------------------
__global__ __launch_bounds__(32)
void attn_kernel(const _Float16* __restrict__ q, const _Float16* __restrict__ k,
                 const _Float16* __restrict__ v, const float* __restrict__ btab,
                 _Float16* __restrict__ ao) {
  __shared__ alignas(16) float    S[64 * 64];
  __shared__ alignas(16) _Float16 P[64 * 64];
  int lane = threadIdx.x;
  int bh = blockIdx.x;
  int h  = bh % NUM_HEADS;
  int b_ = bh / NUM_HEADS;
  const _Float16* qb = q + (size_t)bh * 64 * 32;
  const _Float16* kb = k + (size_t)bh * 64 * 32;
  const _Float16* vb = v + (size_t)bh * 32 * 64;   // transposed: hd x N

  int colL = lane & 15;
  int rsel = (lane >> 4) << 3;

  // S = q @ k^T   (K = hd = 32 -> one wmma per 16x16 tile)
#pragma unroll
  for (int mi = 0; mi < 4; ++mi) {
    v16h a = load_a_frag(qb + mi * 16 * 32, 32, lane);
#pragma unroll
    for (int ni = 0; ni < 4; ++ni) {
      v16h bf = load_b_frag(kb + ni * 16 * 32, 32, lane);
      v8f acc = (v8f){0.f,0.f,0.f,0.f,0.f,0.f,0.f,0.f};
      acc = wmma_f32(a, bf, acc);
#pragma unroll
      for (int e = 0; e < 8; ++e)
        S[(mi * 16 + rsel + e) * 64 + ni * 16 + colL] = acc[e];
    }
  }
  __syncthreads();

  // softmax over rows (2 rows per lane), with rel-pos bias + shift mask.
  int wf = b_ & 511;
  int wd = (wf >> 6) & 7, wh = (wf >> 3) & 7, ww = wf & 7;
#pragma unroll
  for (int rr = 0; rr < 2; ++rr) {
    int i  = lane * 2 + rr;
    int di = i >> 4, hi = (i >> 2) & 3, wi = i & 3;
    int ridI = reg3(wd * 4 + di) * 9 + reg3(wh * 4 + hi) * 3 + reg3(ww * 4 + wi);
    float mx = -1e30f;
    for (int j = 0; j < 64; ++j) {
      int dj = j >> 4, hj = (j >> 2) & 3, wj = j & 3;
      int idx = (di - dj + 3) * 49 + (hi - hj + 3) * 7 + (wi - wj + 3);
      float bv = btab[idx * NUM_HEADS + h];
      int ridJ = reg3(wd * 4 + dj) * 9 + reg3(wh * 4 + hj) * 3 + reg3(ww * 4 + wj);
      float sv = S[i * 64 + j] + bv + ((ridI == ridJ) ? 0.f : -100.f);
      S[i * 64 + j] = sv;
      mx = fmaxf(mx, sv);
    }
    float sum = 0.f;
    for (int j = 0; j < 64; ++j) {
      float e = __expf(S[i * 64 + j] - mx);
      S[i * 64 + j] = e;
      sum += e;
    }
    float inv = 1.f / sum;
    for (int j = 0; j < 64; ++j) P[i * 64 + j] = (_Float16)(S[i * 64 + j] * inv);
  }
  __syncthreads();

  // O = P @ V  (K = 64 tokens -> 2 wmma per tile; V stored hd x N so B loads
  // are contiguous per lane)
#pragma unroll
  for (int mi = 0; mi < 4; ++mi) {
    v16h a0 = load_a_frag(&P[(mi * 16) * 64 + 0],  64, lane);
    v16h a1 = load_a_frag(&P[(mi * 16) * 64 + 32], 64, lane);
#pragma unroll
    for (int ni = 0; ni < 2; ++ni) {
      v16h b0 = load_b_frag(vb + (ni * 16) * 64 + 0,  64, lane);
      v16h b1 = load_b_frag(vb + (ni * 16) * 64 + 32, 64, lane);
      v8f acc = (v8f){0.f,0.f,0.f,0.f,0.f,0.f,0.f,0.f};
      acc = wmma_f32(a0, b0, acc);
      acc = wmma_f32(a1, b1, acc);
#pragma unroll
      for (int e = 0; e < 8; ++e) {
        int row = mi * 16 + rsel + e;
        int col = ni * 16 + colL;
        ao[((size_t)b_ * 64 + row) * C_DIM + h * 32 + col] = (_Float16)acc[e];
      }
    }
  }
}

// ---------------------------------------------------------------------------
extern "C" void kernel_launch(void* const* d_in, const int* in_sizes, int n_in,
                              void* d_out, int out_size, void* d_ws, size_t ws_size,
                              hipStream_t stream) {
  const float* x    = (const float*)d_in[0];
  const float* g1   = (const float*)d_in[1];
  const float* b1   = (const float*)d_in[2];
  const float* Wqkv = (const float*)d_in[3];
  const float* bqkv = (const float*)d_in[4];
  const float* btab = (const float*)d_in[5];
  const float* Wp   = (const float*)d_in[6];
  const float* bp   = (const float*)d_in[7];
  const float* g2   = (const float*)d_in[8];
  const float* b2   = (const float*)d_in[9];
  const float* W1   = (const float*)d_in[10];
  const float* bb1  = (const float*)d_in[11];
  const float* W2   = (const float*)d_in[12];
  const float* bb2  = (const float*)d_in[13];
  float* out = (float*)d_out;

  const size_t M = 65536;   // 2 * 32^3 tokens

  char* ws = (char*)d_ws;
  size_t off = 0;
  auto take = [&](size_t bytes) -> char* {
    char* p = ws + off;
    off += (bytes + 255) & ~(size_t)255;
    return p;
  };
  _Float16* xw16 = (_Float16*)take(M * 384 * 2);    // LN1 windows; reused as h2
  _Float16* q16  = (_Float16*)take(M * 384 * 2);
  _Float16* k16  = (_Float16*)take(M * 384 * 2);
  _Float16* v16b = (_Float16*)take(M * 384 * 2);
  _Float16* ao16 = (_Float16*)take(M * 384 * 2);
  float*    x1   = (float*)   take(M * 384 * 4);    // attn residual output, f32
  _Float16* wq16 = (_Float16*)take(1152 * 384 * 2);
  _Float16* wp16 = (_Float16*)take(384 * 384 * 2);
  _Float16* w116 = (_Float16*)take(1536 * 384 * 2);
  _Float16* w216 = (_Float16*)take(384 * 1536 * 2);
  _Float16* m116 = q16;      // MLP hidden aliases dead q/k/v/ao (exactly 192MB)
  _Float16* h216 = xw16;     // LN2 output aliases dead xw

  // weight conversions f32 -> f16
  int n;
  n = 1152 * 384; f32_to_f16_kernel<<<(n + 255) / 256, 256, 0, stream>>>(Wqkv, wq16, n);
  n = 384 * 384;  f32_to_f16_kernel<<<(n + 255) / 256, 256, 0, stream>>>(Wp,   wp16, n);
  n = 1536 * 384; f32_to_f16_kernel<<<(n + 255) / 256, 256, 0, stream>>>(W1,   w116, n);
  n = 384 * 1536; f32_to_f16_kernel<<<(n + 255) / 256, 256, 0, stream>>>(W2,   w216, n);

  // LN1 + shift + partition
  ln1_shift_partition<<<M / 8, 256, 0, stream>>>(x, g1, b1, xw16);

  // QKV GEMM: (65536 x 384) @ (384 x 1152)
  gemm_wmma_kernel<0><<<dim3(1152 / 64, M / 128), 256, 0, stream>>>(
      xw16, wq16, bqkv, (int)M, 1152, 384, q16, k16, v16b, nullptr, nullptr);

  // windowed attention: 1024 windows * 12 heads
  attn_kernel<<<1024 * NUM_HEADS, 32, 0, stream>>>(q16, k16, v16b, btab, ao16);

  // proj GEMM + reverse partition + roll + shortcut
  gemm_wmma_kernel<1><<<dim3(384 / 64, M / 128), 256, 0, stream>>>(
      ao16, wp16, bp, (int)M, 384, 384, nullptr, nullptr, nullptr, x, x1);

  // LN2
  ln2_kernel<<<M / 8, 256, 0, stream>>>(x1, g2, b2, h216);

  // MLP1 + GELU
  gemm_wmma_kernel<2><<<dim3(1536 / 64, M / 128), 256, 0, stream>>>(
      h216, w116, bb1, (int)M, 1536, 384, m116, nullptr, nullptr, nullptr, nullptr);

  // MLP2 + residual -> out
  gemm_wmma_kernel<3><<<dim3(384 / 64, M / 128), 256, 0, stream>>>(
      m116, w216, bb2, (int)M, 384, 1536, nullptr, nullptr, nullptr, x1, out);
}
